// MultiHeadAttention_76321568850367
// MI455X (gfx1250) — matrix operand order repaired and checked by reading, not compiled, not run on hardware
//
#include <hip/hip_runtime.h>

// ---------------------------------------------------------------------------
// MultiHeadAttention for MI455X (gfx1250, wave32, WMMA bf16 16x16x32)
// Fused flash-attention single pass; scores (537 MB) written exactly once.
// Round-3 changes:
//  * row-max butterfly moved from ds_bpermute (__shfl_xor) to VALU DPP
//    (v_mov_dpp row_ror 1/2/4/8 + v_max) -> zero DS ops for the reduction.
//  * P staging packed as 2xbf16 ds_store_b32 (16 stores vs 32 b16); the
//    implied key-order permutation inside each 64-key block is baked into
//    the Vt layout so P-cols and V-rows agree (scores keep natural order).
//  * global_prefetch of next key block's K/V panels.
// DS per 64-key block: 68 -> 20; VALU reduction work stays on the DPP path.
// ---------------------------------------------------------------------------

typedef __bf16 bf16;
typedef __attribute__((ext_vector_type(8)))  bf16  bf16x8;
typedef __attribute__((ext_vector_type(16))) bf16  bf16x16;
typedef __attribute__((ext_vector_type(8)))  float f32x8;

#define B_   4
#define S_   2048
#define D_   512
#define H_   8
#define DH_  64
#define BH_  (B_ * H_)

__device__ __forceinline__ f32x8 wmma_bf16(bf16x16 a, bf16x16 b, f32x8 c) {
  return __builtin_amdgcn_wmma_f32_16x16x32_bf16(
      /*neg_a=*/false, a, /*neg_b=*/false, b,
      /*c_mod=*/(short)0, c, /*reuse_a=*/false, /*reuse_b=*/false);
}

__device__ __forceinline__ bf16x16 cat8(bf16x8 lo, bf16x8 hi) {
  return __builtin_shufflevector(lo, hi, 0, 1, 2, 3, 4, 5, 6, 7,
                                 8, 9, 10, 11, 12, 13, 14, 15);
}

// Max-reduce across each 16-lane row group using DPP row rotates (pure VALU).
__device__ __forceinline__ float rowmax16_dpp(float x) {
#pragma unroll
  for (int i = 0; i < 4; ++i) {
    const int ctrl = 0x121 + ((1 << i) - 1);  // row_ror:1,2,4,8
    int t;
    switch (i) {  // dpp_ctrl must be a literal constant
      case 0: t = __builtin_amdgcn_mov_dpp(__builtin_bit_cast(int, x), 0x121, 0xf, 0xf, true); break;
      case 1: t = __builtin_amdgcn_mov_dpp(__builtin_bit_cast(int, x), 0x122, 0xf, 0xf, true); break;
      case 2: t = __builtin_amdgcn_mov_dpp(__builtin_bit_cast(int, x), 0x124, 0xf, 0xf, true); break;
      default: t = __builtin_amdgcn_mov_dpp(__builtin_bit_cast(int, x), 0x128, 0xf, 0xf, true); break;
    }
    (void)ctrl;
    x = fmaxf(x, __builtin_bit_cast(float, t));
  }
  return x;
}

__device__ __forceinline__ unsigned pack2_bf16(float lo, float hi) {
  const unsigned a = (unsigned)__builtin_bit_cast(unsigned short, (bf16)lo);
  const unsigned b = (unsigned)__builtin_bit_cast(unsigned short, (bf16)hi);
  return (b << 16) | a;
}

// ---------------------------------------------------------------------------
// Pass 1: scale by diag(w), split heads, cast to bf16.
//   Qbf : [BH][S][DH]  pre-scaled by 1/sqrt(DH)=0.125 (exact pow2 in bf16)
//   Kbf : [BH][S][DH]
//   Vt  : [BH][DH][S]  transposed AND key-permuted within each 64-block to
//         match the packed P staging order: key kk=t*16+li -> column
//         c = 2*li + (t&1) + 32*(t>>1).
// Also copies w to the tail of d_out (third tuple element).
// ---------------------------------------------------------------------------
__global__ __launch_bounds__(256) void prep_kernel(
    const float* __restrict__ q, const float* __restrict__ k,
    const float* __restrict__ v, const float* __restrict__ w,
    bf16* __restrict__ qb, bf16* __restrict__ kb, bf16* __restrict__ vt,
    float* __restrict__ w_out) {
  const int i = blockIdx.x * 256 + threadIdx.x;   // 0 .. B*S*D-1
  const int dcol = i % D_;                        // h*DH + dh
  const int s    = (i / D_) % S_;
  const int b    = i / (D_ * S_);
  const int h    = dcol / DH_;
  const int dh   = dcol % DH_;
  const float ws = w[dcol];
  const int bh   = b * H_ + h;
  const size_t qi = ((size_t)(bh * S_ + s)) * DH_ + dh;

  // key-permuted position for V within its 64-key block
  const int n0 = s & ~63;
  const int kk = s & 63;
  const int t  = kk >> 4;
  const int li = kk & 15;
  const int c  = 2 * li + (t & 1) + (t >> 1) * 32;
  const size_t vi = ((size_t)(bh * DH_ + dh)) * S_ + n0 + c;

  qb[qi] = (bf16)(q[i] * ws * 0.125f);  // fold 1/sqrt(64) into Q
  kb[qi] = (bf16)(k[i] * ws);
  vt[vi] = (bf16)(v[i] * ws);
  if (i < D_) w_out[i] = w[i];
}

// ---------------------------------------------------------------------------
// Pass 2: fused attention. One wave per (bh, 16-query tile), key blocks of 64.
//
// WMMA operand layouts (CDNA5 ISA 7.12.2, 16-bit):
//  A (16x32): lanes 0-15: row M=lane, K={0..7,16..23}; lanes 16-31: row
//             M=lane-16, K={8..15,24..31}  -> two 16B loads per lane.
//  B (32x16): lane j: column j%16, K=(j<16 ? 0..15 : 16..31) contiguous in this kernel (its A loads use the same order).
//  C/D (16x16 f32): VGPR r, lane l -> M = r + (l>=16 ? 8 : 0), N = l%16.
// ---------------------------------------------------------------------------
__global__ __launch_bounds__(256) void attn_kernel(
    const bf16* __restrict__ qbuf, const bf16* __restrict__ kbuf,
    const bf16* __restrict__ vbuf, float* __restrict__ out,
    float* __restrict__ scores) {
  __shared__ __align__(16) bf16 lds_p[8][16][64];  // per-wave P tile, 16 KB

  const int lane = threadIdx.x & 31;
  const int wave = threadIdx.x >> 5;
  // wave-uniform tile id -> force into SGPRs for saddr-based addressing
  const int gw  = __builtin_amdgcn_readfirstlane(blockIdx.x * 8 + wave);
  const int qt  = gw & 127;                 // query tile within sequence
  const int bh  = gw >> 7;                  // batch*head (0..31)
  const int q0  = qt * 16;
  const int l16 = lane & 15;
  const int half = lane >> 4;

  // Uniform (SGPR) base pointers
  const bf16* base_q = qbuf + (size_t)bh * S_ * DH_;
  const bf16* base_k = kbuf + (size_t)bh * S_ * DH_;
  const bf16* base_v = vbuf + (size_t)bh * DH_ * S_;
  float* srow = scores + ((size_t)bh * S_ + (size_t)q0) * S_;

  // --- Q A-fragments (2 chunks of K=32 over DH=64), pre-scaled ------------
  bf16x16 aq[2];
  {
    const int qoff = (q0 + l16) * DH_;
#pragma unroll
    for (int c = 0; c < 2; ++c) {
      const int k0 = qoff + c * 32 + (half ? 16 : 0);
      aq[c] = cat8(*(const bf16x8*)(base_q + k0),
                   *(const bf16x8*)(base_q + k0 + 8));
    }
  }

  // ones B-matrix for WMMA row-sum
  bf16x16 ones;
#pragma unroll
  for (int i = 0; i < 16; ++i) ones[i] = (bf16)1.0f;

  f32x8 o0 = {}, o1 = {}, o2 = {}, o3 = {};
  f32x8 l_acc = {};
  f32x8 m;
#pragma unroll
  for (int r = 0; r < 8; ++r) m[r] = -__builtin_inff();

  for (int blk = 0; blk < S_ / 64; ++blk) {
    const int n0 = blk * 64;                      // scalar
    const bf16* kb_blk = base_k + (size_t)n0 * DH_;
    const bf16* vt_blk = base_v + n0;
    float* srow_blk = srow + n0;

    // prefetch next block's K/V panels (global_prefetch_b8; speculative)
    __builtin_prefetch(kb_blk + 64 * DH_ + (l16 * DH_), 0, 0);
    __builtin_prefetch(vt_blk + 64 + (l16 * S_), 0, 0);

    // --- S = Q K^T for four 16-key tiles (8 WMMA) ------------------------
    f32x8 s[4];
#pragma unroll
    for (int t = 0; t < 4; ++t) {
      bf16x16 bk0, bk1;
      {
        const int koff = (t * 16 + l16) * DH_ + half * 16;
        bk0 = cat8(*(const bf16x8*)(kb_blk + koff),
                   *(const bf16x8*)(kb_blk + koff + 8));
        bk1 = cat8(*(const bf16x8*)(kb_blk + koff + 32),
                   *(const bf16x8*)(kb_blk + koff + 40));
      }
      f32x8 acc = {};
      acc = wmma_bf16(aq[0], bk0, acc);
      s[t] = wmma_bf16(aq[1], bk1, acc);
    }

    // --- stream scores to global (already scaled via Q prescale) ---------
#pragma unroll
    for (int r = 0; r < 8; ++r) {
      float* p = srow_blk + (size_t)(r + half * 8) * S_ + l16;
#pragma unroll
      for (int t = 0; t < 4; ++t) p[t * 16] = s[t][r];
    }

    // --- row max over 64 cols: tile-combine then DPP row reduction -------
    f32x8 mt;
#pragma unroll
    for (int r = 0; r < 8; ++r) {
      mt[r] = fmaxf(fmaxf(s[0][r], s[1][r]), fmaxf(s[2][r], s[3][r]));
      mt[r] = rowmax16_dpp(mt[r]);
    }

    // --- online rescale ---------------------------------------------------
    f32x8 f;
#pragma unroll
    for (int r = 0; r < 8; ++r) {
      const float mn = fmaxf(m[r], mt[r]);
      f[r] = __expf(m[r] - mn);
      m[r] = mn;
    }
#pragma unroll
    for (int r = 0; r < 8; ++r) {
      l_acc[r] *= f[r];
      o0[r] *= f[r];
      o1[r] *= f[r];
      o2[r] *= f[r];
      o3[r] *= f[r];
    }

    // --- P = exp(S - m) packed 2xbf16 -> LDS (key-permuted columns) ------
#pragma unroll
    for (int r = 0; r < 8; ++r) {
      const int row = r + half * 8;
      const float e0 = __expf(s[0][r] - m[r]);
      const float e1 = __expf(s[1][r] - m[r]);
      const float e2 = __expf(s[2][r] - m[r]);
      const float e3 = __expf(s[3][r] - m[r]);
      *(unsigned*)&lds_p[wave][row][2 * l16]      = pack2_bf16(e0, e1);
      *(unsigned*)&lds_p[wave][row][32 + 2 * l16] = pack2_bf16(e2, e3);
    }
    asm volatile("s_wait_dscnt 0" ::: "memory");
    bf16x16 ap[2];
    {
      const bf16* pr = &lds_p[wave][l16][0];
#pragma unroll
      for (int c = 0; c < 2; ++c) {
        const int off = c * 32 + (half ? 16 : 0);
        ap[c] = cat8(*(const bf16x8*)(pr + off),
                     *(const bf16x8*)(pr + off + 8));
      }
    }

    // --- l += rowsum(P) via WMMA against ones (2 WMMA) -------------------
    l_acc = wmma_bf16(ap[0], ones, l_acc);
    l_acc = wmma_bf16(ap[1], ones, l_acc);

    // --- O += P V (8 WMMA over 4 dh tiles x 2 key chunks) ----------------
    // Vt rows are stored in the same permuted key order as the P columns.
#pragma unroll
    for (int t = 0; t < 4; ++t) {
      const int voff = (t * 16 + l16) * S_ + half * 16;
      bf16x16 bv0 = cat8(*(const bf16x8*)(vt_blk + voff),
                         *(const bf16x8*)(vt_blk + voff + 8));
      bf16x16 bv1 = cat8(*(const bf16x8*)(vt_blk + voff + 32),
                         *(const bf16x8*)(vt_blk + voff + 40));
      f32x8* ot = (t == 0) ? &o0 : (t == 1) ? &o1 : (t == 2) ? &o2 : &o3;
      *ot = wmma_bf16(ap[0], bv0, *ot);
      *ot = wmma_bf16(ap[1], bv1, *ot);
    }
  }

  // --- epilogue: O /= l, merge heads, store f32 --------------------------
  const int b = bh / H_;
  const int h = bh % H_;
  float* ob = out + (size_t)b * S_ * D_ + h * DH_;
#pragma unroll
  for (int r = 0; r < 8; ++r) {
    const float rl = 1.0f / l_acc[r];
    const size_t qq = (size_t)(q0 + r + half * 8);
    ob[qq * D_ + 0 * 16 + l16] = o0[r] * rl;
    ob[qq * D_ + 1 * 16 + l16] = o1[r] * rl;
    ob[qq * D_ + 2 * 16 + l16] = o2[r] * rl;
    ob[qq * D_ + 3 * 16 + l16] = o3[r] * rl;
  }
}

// ---------------------------------------------------------------------------
extern "C" void kernel_launch(void* const* d_in, const int* in_sizes, int n_in,
                              void* d_out, int out_size, void* d_ws,
                              size_t ws_size, hipStream_t stream) {
  const float* q = (const float*)d_in[0];
  const float* k = (const float*)d_in[1];
  const float* v = (const float*)d_in[2];
  const float* w = (const float*)d_in[3];

  float* out    = (float*)d_out;                       // [B,S,D]
  float* scores = out + (size_t)B_ * S_ * D_;          // [BH,S,S]
  float* w_out  = scores + (size_t)BH_ * S_ * S_;      // [D]

  // Workspace: 3 bf16 tensors of BH*S*DH = 4M elements each (24 MB total).
  bf16* qb = (bf16*)d_ws;
  bf16* kb = qb + (size_t)BH_ * S_ * DH_;
  bf16* vt = kb + (size_t)BH_ * S_ * DH_;

  const int n = B_ * S_ * D_;  // 4,194,304
  prep_kernel<<<n / 256, 256, 0, stream>>>(q, k, v, w, qb, kb, vt, w_out);

  // 4096 waves: one per (bh, 16-query tile); 8 waves per 256-thread WG.
  attn_kernel<<<4096 / 8, 256, 0, stream>>>(qb, kb, vt, out, scores);
}
